// Matformer_62405874811572
// MI455X (gfx1250) — matrix operand order repaired and checked by reading, not compiled, not run on hardware
//
#include <hip/hip_runtime.h>
#include <math.h>

// ---------------- problem constants ----------------
constexpr int N = 16000, E = 128000, G = 64;
constexpr int D = 64, H = 4, O = 64, L = 5, HID = 128;
constexpr float EPS = 1e-5f;
constexpr float SCALE = 0.072168783648703332f; // 1/sqrt(192)

// ---------------- vector types ----------------
typedef __attribute__((ext_vector_type(16))) _Float16 v16h;
typedef __attribute__((ext_vector_type(8)))  _Float16 v8h;
typedef __attribute__((ext_vector_type(4)))  _Float16 v4h;
typedef __attribute__((ext_vector_type(8)))  float    v8f;

__device__ __forceinline__ v8f wmma_f16(v16h a, v16h b, v8f c) {
  return __builtin_amdgcn_wmma_f32_16x16x32_f16(false, a, false, b, (short)0, c,
                                                false, false);
}

// Build an A-fragment (16x32 f16 tile, wave32 layout) from a row-major row
// pointer (LDS or global).  Lane m = lane&15 owns row m; lanes 0-15 read
// K = kb+{0..7, 16..23}, lanes 16-31 read K = kb+{8..15, 24..31}.
// Both halves are contiguous 8-half (16B) runs -> two b128 loads.
__device__ __forceinline__ v16h afrag_from(const _Float16* rowPtr, int kb, int lane) {
  int kh = (lane >> 4) * 8;
  v8h lo = *(const v8h*)(rowPtr + kb + kh);
  v8h hi = *(const v8h*)(rowPtr + kb + 16 + kh);
  return __builtin_shufflevector(lo, hi, 0,1,2,3,4,5,6,7,8,9,10,11,12,13,14,15);
}

__device__ __forceinline__ float sigf(float x) { return 1.f / (1.f + expf(-x)); }

// ---------------- weight pre-pack: row-major f32 -> B-fragment f16 ----------------
// Consumption: v16h b = ((const v16h*)pW)[(kt*ntiles + nt)*32 + lane];
// b[j] = W[kt*32 + (j>>3)*16 + (lane>>4)*8 + (j&7)][nt*16 + (lane&15)]   (ISA B layout)
__global__ void pack_b_kernel(const float* __restrict__ W, _Float16* __restrict__ dst,
                              int K, int Nc) {
  int tid = blockIdx.x * blockDim.x + threadIdx.x;
  int total = K * Nc;
  if (tid >= total) return;
  int ntiles = Nc / 16;
  int f = tid / 512, r = tid % 512;
  int lane = r / 16, j = r % 16;
  int kt = f / ntiles, nt = f % ntiles;
  int k = kt * 32 + (j >> 3) * 16 + (lane >> 4) * 8 + (j & 7);
  int n = nt * 16 + (lane & 15);
  dst[tid] = (_Float16)W[(size_t)k * Nc + n];
}

// ---------------- h = x @ W_atom + b_atom ----------------
__global__ void h_init_kernel(const float* __restrict__ x, const float* __restrict__ Wa,
                              const float* __restrict__ ba, float* __restrict__ h) {
  int t = blockIdx.x * blockDim.x + threadIdx.x;
  if (t >= N * 64) return;
  int nr = t >> 6, c = t & 63;
  float s = ba[c];
#pragma unroll
  for (int j = 0; j < 4; ++j) s += x[nr * 4 + j] * Wa[j * 64 + c];
  h[t] = s;
}

// ---------------- e = softplus64(edge_attr@We1+be1) @ We2 + be2 -> f16 [E,64] ----------------
__global__ __launch_bounds__(32) void edge_embed_kernel(
    const float* __restrict__ ea, const float* __restrict__ We1,
    const float* __restrict__ be1, const _Float16* __restrict__ pWe2,
    const float* __restrict__ be2, _Float16* __restrict__ ef) {
  __shared__ __align__(16) _Float16 tb[16][64];
  const int lane = threadIdx.x;
  const int e0 = blockIdx.x * 16;
  for (int i = lane; i < 16 * 64; i += 32) {
    int r = i >> 6, c = i & 63;
    float s = be1[c];
#pragma unroll
    for (int j = 0; j < 9; ++j) s += ea[(size_t)(e0 + r) * 9 + j] * We1[j * 64 + c];
    float z = s * 64.f;
    float y = (z > 64.f) ? s : log1pf(expf(fminf(z, 64.f))) * (1.f / 64.f);
    tb[r][c] = (_Float16)y;
  }
  __syncthreads();
  const int n = lane & 15, mOff = (lane >> 4) * 8;
  v16h a0 = afrag_from(&tb[n][0], 0, lane);
  v16h a1 = afrag_from(&tb[n][0], 32, lane);
  const v16h* pB = (const v16h*)pWe2;
#pragma unroll
  for (int nt = 0; nt < 4; ++nt) {
    float bb = be2[nt * 16 + n];
    v8f acc;
#pragma unroll
    for (int i = 0; i < 8; ++i) acc[i] = bb;
    acc = wmma_f16(a0, pB[(0 * 4 + nt) * 32 + lane], acc);
    acc = wmma_f16(a1, pB[(1 * 4 + nt) * 32 + lane], acc);
#pragma unroll
    for (int r = 0; r < 8; ++r)
      ef[(size_t)(e0 + mOff + r) * 64 + nt * 16 + n] = (_Float16)acc[r];
  }
}

// ---------------- node projections: Q,K,V (f16 [N,256]) and fea (f32 [N,64]) ----------------
__global__ __launch_bounds__(32) void node_proj_kernel(
    const float* __restrict__ h,
    const _Float16* __restrict__ pWq, const float* __restrict__ bq,
    const _Float16* __restrict__ pWk, const float* __restrict__ bk,
    const _Float16* __restrict__ pWv, const float* __restrict__ bv,
    const _Float16* __restrict__ pWfea, const float* __restrict__ bfea,
    _Float16* __restrict__ Qf, _Float16* __restrict__ Kf, _Float16* __restrict__ Vf,
    float* __restrict__ fea) {
  __shared__ __align__(16) _Float16 hb[16][64];
  const int lane = threadIdx.x;
  const int t0 = blockIdx.x * 16;
  // 16*64 floats = 256 float4; 8 per lane
  for (int i = lane; i < 256; i += 32) {
    int r = i >> 4, c4 = (i & 15) * 4;
    float4 f = *(const float4*)(h + (size_t)(t0 + r) * 64 + c4);
    v4h hv; hv[0] = (_Float16)f.x; hv[1] = (_Float16)f.y;
    hv[2] = (_Float16)f.z; hv[3] = (_Float16)f.w;
    *(v4h*)(&hb[r][c4]) = hv;
  }
  __syncthreads();
  const int n = lane & 15, mOff = (lane >> 4) * 8;
  v16h a0 = afrag_from(&hb[n][0], 0, lane);
  v16h a1 = afrag_from(&hb[n][0], 32, lane);

  const _Float16* pws[3] = {pWq, pWk, pWv};
  const float* bs[3] = {bq, bk, bv};
  _Float16* outs[3] = {Qf, Kf, Vf};
#pragma unroll
  for (int w = 0; w < 3; ++w) {
    const v16h* pB = (const v16h*)pws[w];
    for (int nt = 0; nt < 16; ++nt) {
      float bb = bs[w][nt * 16 + n];
      v8f acc;
#pragma unroll
      for (int i = 0; i < 8; ++i) acc[i] = bb;
      acc = wmma_f16(a0, pB[(0 * 16 + nt) * 32 + lane], acc);
      acc = wmma_f16(a1, pB[(1 * 16 + nt) * 32 + lane], acc);
#pragma unroll
      for (int r = 0; r < 8; ++r)
        outs[w][(size_t)(t0 + mOff + r) * 256 + nt * 16 + n] = (_Float16)acc[r];
    }
  }
  const v16h* pB = (const v16h*)pWfea;
#pragma unroll
  for (int nt = 0; nt < 4; ++nt) {
    float bb = bfea[nt * 16 + n];
    v8f acc;
#pragma unroll
    for (int i = 0; i < 8; ++i) acc[i] = bb;
    acc = wmma_f16(a0, pB[(0 * 4 + nt) * 32 + lane], acc);
    acc = wmma_f16(a1, pB[(1 * 4 + nt) * 32 + lane], acc);
#pragma unroll
    for (int r = 0; r < 8; ++r)
      fea[(size_t)(t0 + mOff + r) * 64 + nt * 16 + n] = acc[r];
  }
}

// ---------------- the big fused per-edge kernel ----------------
// grid = (E/16, H); one wave per block; 16 edges x 1 head per tile.
// LDS: abuf (v_ij, then reused for gated m1) 6KB + gbuf 12KB + ehbuf 2KB +
//      m2buf 4KB = 24KB -> 13 blocks per 320KB WGP.
__global__ __launch_bounds__(32) void edge_att_kernel(
    const int* __restrict__ srcI, const int* __restrict__ dstI,
    const _Float16* __restrict__ Qf, const _Float16* __restrict__ Kf,
    const _Float16* __restrict__ Vf, const _Float16* __restrict__ ef,
    const _Float16* __restrict__ pWedge, const float* __restrict__ bedge,
    const _Float16* __restrict__ pWupd, const float* __restrict__ bupd,
    const _Float16* __restrict__ pWmsg, const float* __restrict__ bmsg,
    const float* __restrict__ g_ln, const float* __restrict__ be_ln,
    const float* __restrict__ g_msg, const float* __restrict__ be_msg,
    float* __restrict__ agg) {
  __shared__ __align__(16) _Float16 abuf[16][192];   // v_ij, then gate*u
  __shared__ __align__(16) float    gbuf[16][192];   // alpha -> gate
  __shared__ __align__(16) _Float16 ehbuf[16][64];   // Eh tile (f16)
  __shared__ __align__(16) float    m2buf[16][64];   // pre-LN message
  const int lane = threadIdx.x;
  const int et = blockIdx.x;
  const int hh = blockIdx.y;
  const int n = lane & 15, mOff = (lane >> 4) * 8;

  if (lane == 0 && et + 1 < gridDim.x) __builtin_prefetch(srcI + (et + 1) * 16, 0, 0);

  // (a) Eh tile = e[16x64] @ Wedge[:, hh*64 : hh*64+64]  (WMMA, K=64)
  {
    const _Float16* erow = ef + (size_t)(et * 16 + n) * 64;
    v16h a0 = afrag_from(erow, 0, lane);
    v16h a1 = afrag_from(erow, 32, lane);
    const v16h* pB = (const v16h*)pWedge;
#pragma unroll
    for (int nt = 0; nt < 4; ++nt) {
      int ng = hh * 4 + nt;
      float bb = bedge[hh * 64 + nt * 16 + n];
      v8f acc;
#pragma unroll
      for (int i = 0; i < 8; ++i) acc[i] = bb;
      acc = wmma_f16(a0, pB[(0 * 16 + ng) * 32 + lane], acc);
      acc = wmma_f16(a1, pB[(1 * 16 + ng) * 32 + lane], acc);
#pragma unroll
      for (int r = 0; r < 8; ++r) ehbuf[mOff + r][nt * 16 + n] = (_Float16)acc[r];
    }
  }
  __syncthreads();

  // (b) alpha -> LN(192) -> sigmoid gate ; build v_ij (f16).  2 lanes/row,
  //     12 aligned 8-half chunks per lane (vector b128 gathers, L2-resident).
  {
    const int r = lane & 15, half = lane >> 4;
    const int e = et * 16 + r;
    const int si = srcI[e], di = dstI[e];
    const _Float16* Qr = Qf + (size_t)di * 256 + hh * 64;
    const _Float16* Kd = Kf + (size_t)di * 256 + hh * 64;
    const _Float16* Ks = Kf + (size_t)si * 256 + hh * 64;
    const _Float16* Vd = Vf + (size_t)di * 256 + hh * 64;
    const _Float16* Vs = Vf + (size_t)si * 256 + hh * 64;
    float sum = 0.f, sq = 0.f;
#pragma unroll
    for (int ch = 0; ch < 12; ++ch) {
      int c0 = half * 96 + ch * 8;
      int seg = c0 >> 6, cc0 = c0 & 63;
      v8h q8 = *(const v8h*)(Qr + cc0);
      v8h k8, v8;
      if (seg == 0)      { k8 = *(const v8h*)(Kd + cc0); v8 = *(const v8h*)(Vd + cc0); }
      else if (seg == 1) { k8 = *(const v8h*)(Ks + cc0); v8 = *(const v8h*)(Vs + cc0); }
      else               { k8 = *(const v8h*)(&ehbuf[r][cc0]); v8 = k8; }
      *(v8h*)(&abuf[r][c0]) = v8;
#pragma unroll
      for (int i = 0; i < 8; ++i) {
        float al = (float)q8[i] * (float)k8[i] * SCALE;
        gbuf[r][c0 + i] = al;
        sum += al; sq += al * al;
      }
    }
    sum += __shfl_xor(sum, 16, 32);
    sq  += __shfl_xor(sq, 16, 32);
    float mu = sum * (1.f / 192.f);
    float var = sq * (1.f / 192.f) - mu * mu;
    float rstd = rsqrtf(var + EPS);
    for (int c = half * 96; c < half * 96 + 96; ++c) {
      float nrm = (gbuf[r][c] - mu) * rstd * g_ln[c] + be_ln[c];
      gbuf[r][c] = sigf(nrm);
    }
  }
  __syncthreads();

  // (d) u = v_ij @ Wupd + bupd ; m1 = gate * u  (WMMA, K=192, 12 col tiles)
  //     A-fragments fully register-resident before abuf is overwritten by m1.
  {
    v16h aF[6];
#pragma unroll
    for (int kt = 0; kt < 6; ++kt) aF[kt] = afrag_from(&abuf[n][0], kt * 32, lane);
    __syncthreads();
    const v16h* pB = (const v16h*)pWupd;
    for (int nt = 0; nt < 12; ++nt) {
      int col = nt * 16 + n;
      float bb = bupd[col];
      v8f acc;
#pragma unroll
      for (int i = 0; i < 8; ++i) acc[i] = bb;
#pragma unroll
      for (int kt = 0; kt < 6; ++kt)
        acc = wmma_f16(aF[kt], pB[(kt * 12 + nt) * 32 + lane], acc);
#pragma unroll
      for (int r = 0; r < 8; ++r)
        abuf[mOff + r][col] = (_Float16)(acc[r] * gbuf[mOff + r][col]);
    }
  }
  __syncthreads();

  // (f) m2 = m1 @ Wmsg + bmsg  (WMMA, K=192, 4 col tiles)
  {
    v16h aF[6];
#pragma unroll
    for (int kt = 0; kt < 6; ++kt) aF[kt] = afrag_from(&abuf[n][0], kt * 32, lane);
    const v16h* pB = (const v16h*)pWmsg;
#pragma unroll
    for (int nt = 0; nt < 4; ++nt) {
      int col = nt * 16 + n;
      float bb = bmsg[col];
      v8f acc;
#pragma unroll
      for (int i = 0; i < 8; ++i) acc[i] = bb;
#pragma unroll
      for (int kt = 0; kt < 6; ++kt)
        acc = wmma_f16(aF[kt], pB[(kt * 4 + nt) * 32 + lane], acc);
#pragma unroll
      for (int r = 0; r < 8; ++r) m2buf[mOff + r][col] = acc[r];
    }
  }
  __syncthreads();

  // (g) LN(64) per row, scatter-add into agg[dst, hh*64 + c]
  {
    const int r = lane & 15, half = lane >> 4;
    float s = 0.f, q = 0.f;
    for (int c = half * 32; c < half * 32 + 32; ++c) {
      float v = m2buf[r][c];
      s += v; q += v * v;
    }
    s += __shfl_xor(s, 16, 32);
    q += __shfl_xor(q, 16, 32);
    float mu = s * (1.f / 64.f);
    float var = q * (1.f / 64.f) - mu * mu;
    float rstd = rsqrtf(var + EPS);
    const int e = et * 16 + r;
    float* aggRow = agg + (size_t)dstI[e] * 256 + hh * 64;
    for (int c = half * 32; c < half * 32 + 32; ++c) {
      float v = (m2buf[r][c] - mu) * rstd * g_msg[c] + be_msg[c];
      atomicAdd(&aggRow[c], v);
    }
  }
}

// ---------------- out_pre = agg @ Wcat + bcat  (WMMA, K=256) ----------------
__global__ __launch_bounds__(32) void node_update_kernel(
    const float* __restrict__ agg, const _Float16* __restrict__ pWcat,
    const float* __restrict__ bcat, float* __restrict__ outp) {
  __shared__ __align__(16) _Float16 ab[16][256];
  const int lane = threadIdx.x;
  const int t0 = blockIdx.x * 16;
  // 16*256 floats = 1024 float4; 32 per lane
  for (int i = lane; i < 1024; i += 32) {
    int r = i >> 6, c4 = (i & 63) * 4;
    float4 f = *(const float4*)(agg + (size_t)(t0 + r) * 256 + c4);
    v4h hv; hv[0] = (_Float16)f.x; hv[1] = (_Float16)f.y;
    hv[2] = (_Float16)f.z; hv[3] = (_Float16)f.w;
    *(v4h*)(&ab[r][c4]) = hv;
  }
  __syncthreads();
  const int n = lane & 15, mOff = (lane >> 4) * 8;
  v16h aF[8];
#pragma unroll
  for (int kt = 0; kt < 8; ++kt) aF[kt] = afrag_from(&ab[n][0], kt * 32, lane);
  const v16h* pB = (const v16h*)pWcat;
#pragma unroll
  for (int nt = 0; nt < 4; ++nt) {
    float bb = bcat[nt * 16 + n];
    v8f acc;
#pragma unroll
    for (int i = 0; i < 8; ++i) acc[i] = bb;
#pragma unroll
    for (int kt = 0; kt < 8; ++kt)
      acc = wmma_f16(aF[kt], pB[(kt * 4 + nt) * 32 + lane], acc);
#pragma unroll
    for (int r = 0; r < 8; ++r)
      outp[(size_t)(t0 + mOff + r) * 64 + nt * 16 + n] = acc[r];
  }
}

// ---------------- BN stats / apply ----------------
__global__ void bn_reduce_kernel(const float* __restrict__ outp, float* __restrict__ bn) {
  int c = threadIdx.x;           // 64 channels
  int r0 = blockIdx.x * 160;     // 100 blocks * 160 rows = 16000
  float s = 0.f, q = 0.f;
  for (int i = 0; i < 160; ++i) {
    float v = outp[(size_t)(r0 + i) * 64 + c];
    s += v; q += v * v;
  }
  atomicAdd(&bn[c], s);
  atomicAdd(&bn[64 + c], q);
}

__global__ void bn_apply_kernel(const float* __restrict__ outp, const float* __restrict__ bn,
                                const float* __restrict__ g_bn, const float* __restrict__ be_bn,
                                const float* __restrict__ fea, float* __restrict__ h) {
  int t = blockIdx.x * blockDim.x + threadIdx.x;
  if (t >= N * 64) return;
  int c = t & 63;
  float mu = bn[c] * (1.f / (float)N);
  float var = bn[64 + c] * (1.f / (float)N) - mu * mu;
  float v = (outp[t] - mu) * rsqrtf(var + EPS) * g_bn[c] + be_bn[c];
  h[t] = v * sigf(v) + fea[t];
}

// ---------------- pooling + final MLP ----------------
__global__ void zero_kernel(float* p, int n) {
  int t = blockIdx.x * blockDim.x + threadIdx.x;
  if (t < n) p[t] = 0.f;
}
__global__ void pool_cnt_kernel(const int* __restrict__ batch, float* __restrict__ cnt) {
  int t = blockIdx.x * blockDim.x + threadIdx.x;
  if (t < N) atomicAdd(&cnt[batch[t]], 1.f);
}
__global__ void pool_sum_kernel(const int* __restrict__ batch, const float* __restrict__ h,
                                float* __restrict__ pooled) {
  int t = blockIdx.x * blockDim.x + threadIdx.x;
  if (t >= N * 64) return;
  int nr = t >> 6, c = t & 63;
  atomicAdd(&pooled[(size_t)batch[nr] * 64 + c], h[t]);
}
__global__ __launch_bounds__(128) void final_kernel(
    const float* __restrict__ pooled, const float* __restrict__ cnt,
    const float* __restrict__ Wfc, const float* __restrict__ bfc,
    const float* __restrict__ Wout, const float* __restrict__ bout,
    float* __restrict__ out) {
  __shared__ float zs[G][HID];
  int j = threadIdx.x;  // 128
  for (int g = 0; g < G; ++g) {
    float ic = 1.f / fmaxf(cnt[g], 1.f);
    float s = bfc[j];
    for (int c = 0; c < 64; ++c) s += pooled[g * 64 + c] * ic * Wfc[c * HID + j];
    zs[g][j] = s * sigf(s);
  }
  __syncthreads();
  if (j < G) {
    float s = bout[0];
    for (int k = 0; k < HID; ++k) s += zs[j][k] * Wout[k];
    out[j] = s;
  }
}

// ---------------- workspace layout ----------------
constexpr size_t OFF_H    = 0;
constexpr size_t OFF_EF   = OFF_H   + (size_t)N * 64 * 4;      // h f32
constexpr size_t OFF_Q    = OFF_EF  + (size_t)E * 64 * 2;      // e f16
constexpr size_t OFF_K    = OFF_Q   + (size_t)N * 256 * 2;
constexpr size_t OFF_V    = OFF_K   + (size_t)N * 256 * 2;
constexpr size_t OFF_FEA  = OFF_V   + (size_t)N * 256 * 2;
constexpr size_t OFF_AGG  = OFF_FEA + (size_t)N * 64 * 4;
constexpr size_t OFF_OUTP = OFF_AGG + (size_t)N * 256 * 4;
constexpr size_t OFF_BN   = OFF_OUTP + (size_t)N * 64 * 4;
constexpr size_t OFF_POOL = OFF_BN + 512;
constexpr size_t OFF_CNT  = OFF_POOL + (size_t)G * 64 * 4;
constexpr size_t OFF_PWE2 = OFF_CNT + 256;
constexpr size_t OFF_PW   = OFF_PWE2 + 8192;
// packed per-layer block (halfs):
constexpr size_t PL_Q = 0, PL_K = 16384, PL_V = 32768, PL_EDGE = 49152,
                 PL_UPD = 65536, PL_MSG = 102400, PL_CAT = 114688,
                 PL_FEA = 131072, PL_TOTAL = 135168;

extern "C" void kernel_launch(void* const* d_in, const int* in_sizes, int n_in,
                              void* d_out, int out_size, void* d_ws, size_t ws_size,
                              hipStream_t stream) {
  const float* x         = (const float*)d_in[0];
  const float* edge_attr = (const float*)d_in[1];
  const int*   eidx      = (const int*)d_in[2];
  const int*   batch     = (const int*)d_in[3];
  const float* W_atom = (const float*)d_in[4];  const float* b_atom = (const float*)d_in[5];
  const float* We1    = (const float*)d_in[6];  const float* be1    = (const float*)d_in[7];
  const float* We2    = (const float*)d_in[8];  const float* be2    = (const float*)d_in[9];
  const float* Wq  = (const float*)d_in[10]; const float* bq  = (const float*)d_in[11];
  const float* Wk  = (const float*)d_in[12]; const float* bk  = (const float*)d_in[13];
  const float* Wv  = (const float*)d_in[14]; const float* bv  = (const float*)d_in[15];
  const float* Wed = (const float*)d_in[16]; const float* bed = (const float*)d_in[17];
  const float* Wfe = (const float*)d_in[18]; const float* bfe = (const float*)d_in[19];
  const float* Wct = (const float*)d_in[20]; const float* bct = (const float*)d_in[21];
  const float* Wup = (const float*)d_in[22]; const float* bup = (const float*)d_in[23];
  const float* Wms = (const float*)d_in[24]; const float* bms = (const float*)d_in[25];
  const float* gms = (const float*)d_in[26]; const float* bems= (const float*)d_in[27];
  const float* gln = (const float*)d_in[28]; const float* beln= (const float*)d_in[29];
  const float* gbn = (const float*)d_in[30]; const float* bebn= (const float*)d_in[31];
  const float* Wfc = (const float*)d_in[32]; const float* bfc = (const float*)d_in[33];
  const float* Wou = (const float*)d_in[34]; const float* bou = (const float*)d_in[35];

  char* ws = (char*)d_ws;
  float*    hbuf  = (float*)(ws + OFF_H);
  _Float16* ef    = (_Float16*)(ws + OFF_EF);
  _Float16* Qf    = (_Float16*)(ws + OFF_Q);
  _Float16* Kf    = (_Float16*)(ws + OFF_K);
  _Float16* Vf    = (_Float16*)(ws + OFF_V);
  float*    fea   = (float*)(ws + OFF_FEA);
  float*    agg   = (float*)(ws + OFF_AGG);
  float*    outp  = (float*)(ws + OFF_OUTP);
  float*    bn    = (float*)(ws + OFF_BN);
  float*    pooled= (float*)(ws + OFF_POOL);
  float*    cnt   = (float*)(ws + OFF_CNT);
  _Float16* pWe2  = (_Float16*)(ws + OFF_PWE2);
  _Float16* pw    = (_Float16*)(ws + OFF_PW);

  const int* srcI = eidx;
  const int* dstI = eidx + E;

  auto pack = [&](const float* W, _Float16* dst, int K, int Nc) {
    int tot = K * Nc;
    pack_b_kernel<<<(tot + 255) / 256, 256, 0, stream>>>(W, dst, K, Nc);
  };

  // --- weight pre-pack into WMMA B-fragment layout (f16) ---
  pack(We2, pWe2, 64, 64);
  for (int l = 0; l < L; ++l) {
    _Float16* base = pw + (size_t)l * PL_TOTAL;
    pack(Wq  + (size_t)l * 64 * 256, base + PL_Q,    64, 256);
    pack(Wk  + (size_t)l * 64 * 256, base + PL_K,    64, 256);
    pack(Wv  + (size_t)l * 64 * 256, base + PL_V,    64, 256);
    pack(Wed + (size_t)l * 64 * 256, base + PL_EDGE, 64, 256);
    pack(Wup + (size_t)l * 192 * 192, base + PL_UPD, 192, 192);
    pack(Wms + (size_t)l * 192 * 64,  base + PL_MSG, 192, 64);
    pack(Wct + (size_t)l * 256 * 64,  base + PL_CAT, 256, 64);
    pack(Wfe + (size_t)l * 64 * 64,   base + PL_FEA, 64, 64);
  }

  // --- embeddings ---
  h_init_kernel<<<(N * 64 + 255) / 256, 256, 0, stream>>>(x, W_atom, b_atom, hbuf);
  edge_embed_kernel<<<E / 16, 32, 0, stream>>>(edge_attr, We1, be1, pWe2, be2, ef);

  // --- layers ---
  for (int l = 0; l < L; ++l) {
    _Float16* base = pw + (size_t)l * PL_TOTAL;
    node_proj_kernel<<<N / 16, 32, 0, stream>>>(
        hbuf,
        base + PL_Q, bq + (size_t)l * 256,
        base + PL_K, bk + (size_t)l * 256,
        base + PL_V, bv + (size_t)l * 256,
        base + PL_FEA, bfe + (size_t)l * 64,
        Qf, Kf, Vf, fea);
    zero_kernel<<<(N * 256 + 255) / 256, 256, 0, stream>>>(agg, N * 256);
    zero_kernel<<<1, 128, 0, stream>>>(bn, 128);
    edge_att_kernel<<<dim3(E / 16, H), 32, 0, stream>>>(
        srcI, dstI, Qf, Kf, Vf, ef,
        base + PL_EDGE, bed + (size_t)l * 256,
        base + PL_UPD,  bup + (size_t)l * 192,
        base + PL_MSG,  bms + (size_t)l * 64,
        gln + (size_t)l * 192, beln + (size_t)l * 192,
        gms + (size_t)l * 64,  bems + (size_t)l * 64,
        agg);
    node_update_kernel<<<N / 16, 32, 0, stream>>>(agg, base + PL_CAT,
                                                  bct + (size_t)l * 64, outp);
    bn_reduce_kernel<<<100, 64, 0, stream>>>(outp, bn);
    bn_apply_kernel<<<(N * 64 + 255) / 256, 256, 0, stream>>>(
        outp, bn, gbn + (size_t)l * 64, bebn + (size_t)l * 64, fea, hbuf);
  }

  // --- pooling + readout ---
  zero_kernel<<<(G * 64 + 255) / 256, 256, 0, stream>>>(pooled, G * 64);
  zero_kernel<<<1, G, 0, stream>>>(cnt, G);
  pool_cnt_kernel<<<(N + 255) / 256, 256, 0, stream>>>(batch, cnt);
  pool_sum_kernel<<<(N * 64 + 255) / 256, 256, 0, stream>>>(batch, hbuf, pooled);
  final_kernel<<<1, 128, 0, stream>>>(pooled, cnt, Wfc, bfc, Wou, bou, (float*)d_out);
}
